// SpatialAttention2D_36567351558908
// MI455X (gfx1250) — compile-verified, matrix operand order checked by reading, not run on hardware
//
#include <hip/hip_runtime.h>

// ---------------------------------------------------------------------------
// CDNA5 (gfx1250) wave32 WMMA implementation of 2D-RoPE multi-head attention.
// All GEMMs use v_wmma_f32_16x16x32_bf16 (fp32 accumulate, bf16 operands).
// Kernel 2 stages Q via the Tensor Data Mover (tensor_load_to_lds).
// ---------------------------------------------------------------------------

typedef __attribute__((ext_vector_type(16))) __bf16       v16bf;
typedef __attribute__((ext_vector_type(8)))  float        v8f;
typedef __attribute__((ext_vector_type(4)))  unsigned int v4u;
typedef __attribute__((ext_vector_type(8)))  unsigned int v8u;

#define WMMA_BF16(A, B, C) \
  __builtin_amdgcn_wmma_f32_16x16x32_bf16(false, (A), false, (B), (short)0, (C), false, false)

// K offset of fragment element e for lane-group g (lane>>4), per ISA 16-bit
// A/B layout: lanes 0-15 hold K {0..7,16..23}, lanes 16-31 hold K {8..15,24..31}.
__device__ __forceinline__ int fk(int e, int g) {
  return ((e < 8) ? e : (e + 8)) + (g << 3);
}

// Fragment from row-major f32 source (rows along M/N, contiguous K), cvt->bf16.
__device__ __forceinline__ v16bf frag_rm_f32(const float* __restrict__ p, int ld,
                                             int r0, int k0, int lane) {
  const float* row = p + (long)(r0 + (lane & 15)) * ld + k0;
  int g = lane >> 4;
  v16bf f;
#pragma unroll
  for (int e = 0; e < 16; ++e) f[e] = (__bf16)row[fk(e, g)];
  return f;
}

// Fragment from row-major bf16 source (works for global or LDS pointers).
__device__ __forceinline__ v16bf frag_rm_bf16(const __bf16* __restrict__ p, int ld,
                                              int r0, int k0, int lane) {
  const __bf16* row = p + (long)(r0 + (lane & 15)) * ld + k0;
  int g = lane >> 4;
  v16bf f;
#pragma unroll
  for (int e = 0; e < 16; ++e) f[e] = row[fk(e, g)];
  return f;
}

// Row-guarded row-major bf16 fragment (rows >= rmax read as zero).
__device__ __forceinline__ v16bf frag_rm_bf16_guard(const __bf16* __restrict__ p, int ld,
                                                    int r0, int k0, int rmax, int lane) {
  int r = r0 + (lane & 15);
  int g = lane >> 4;
  v16bf f;
  if (r < rmax) {
    const __bf16* row = p + (long)r * ld + k0;
#pragma unroll
    for (int e = 0; e < 16; ++e) f[e] = row[fk(e, g)];
  } else {
#pragma unroll
    for (int e = 0; e < 16; ++e) f[e] = (__bf16)0.0f;
  }
  return f;
}

// B fragment from a K-major (K x N row-major) bf16 source, K-guarded.
__device__ __forceinline__ v16bf frag_kn_bf16_guard(const __bf16* __restrict__ p, int ld,
                                                    int k0, int n0, int kmax, int lane) {
  int n = n0 + (lane & 15);
  int g = lane >> 4;
  v16bf f;
#pragma unroll
  for (int e = 0; e < 16; ++e) {
    int k = k0 + fk(e, g);
    f[e] = (k < kmax) ? p[(long)k * ld + n] : (__bf16)0.0f;
  }
  return f;
}

// ---------------------------------------------------------------------------
// Kernel 1: qkv = x @ qkv_w^T + qkv_b, fused 2D-RoPE on q,k, q *= 1/sqrt(64).
// One wave computes one full head slice (16 tokens x 64 dims) of one section:
// 4 accumulators (d0 = 0,16,32,48) share one A fragment; RoPE partners
// (d, d+32) are register-local. Output bf16 [B*H, 196, 64].
// ---------------------------------------------------------------------------
__global__ void __launch_bounds__(256) qkv_rope_kernel(
    const float* __restrict__ x, const float* __restrict__ qkv_w,
    const float* __restrict__ qkv_b,
    __bf16* __restrict__ qb, __bf16* __restrict__ kb, __bf16* __restrict__ vb)
{
  int lane = threadIdx.x & 31;
  int wid  = blockIdx.x * (blockDim.x >> 5) + (threadIdx.x >> 5);
  int mt   = wid / 36;           // token tile: 25088/16 = 1568
  int rem  = wid % 36;           // 3 sections * 12 heads
  int s    = rem / 12;           // 0=q 1=k 2=v
  int h    = rem % 12;
  int cb   = s * 768 + h * 64;   // first output column of this head slice
  int m0   = mt * 16;

  v8f ac0 = {}, ac1 = {}, ac2 = {}, ac3 = {};
#pragma unroll 1
  for (int kk = 0; kk < 768; kk += 32) {
    if (kk + 32 < 768) {   // prefetch next K tile (global_prefetch_b8)
      __builtin_prefetch(x     + (long)(m0 + (lane & 15)) * 768 + kk + 32, 0, 3);
      __builtin_prefetch(qkv_w + (long)(cb + (lane & 15)) * 768 + kk + 32, 0, 3);
    }
    v16bf a  = frag_rm_f32(x,     768, m0,      kk, lane);
    v16bf b0 = frag_rm_f32(qkv_w, 768, cb,      kk, lane);
    v16bf b1 = frag_rm_f32(qkv_w, 768, cb + 16, kk, lane);
    v16bf b2 = frag_rm_f32(qkv_w, 768, cb + 32, kk, lane);
    v16bf b3 = frag_rm_f32(qkv_w, 768, cb + 48, kk, lane);
    ac0 = WMMA_BF16(a, b0, ac0);
    ac1 = WMMA_BF16(a, b1, ac1);
    ac2 = WMMA_BF16(a, b2, ac2);
    ac3 = WMMA_BF16(a, b3, ac3);
  }

  int   nn = lane & 15;
  int   g  = lane >> 4;
  float bi0 = qkv_b[cb + nn],      bi1 = qkv_b[cb + 16 + nn];
  float bi2 = qkv_b[cb + 32 + nn], bi3 = qkv_b[cb + 48 + nn];
  // inv_freq = 10000^(-nn/16) = exp2(-nn * log2(1e4)/16); single v_exp_f32
  float invf   = __builtin_amdgcn_exp2f(-(float)nn * 0.830482024f);
  float qscale = (s == 0) ? 0.125f : 1.0f;     // 1/sqrt(head_dim)
  __bf16* dst  = (s == 0) ? qb : (s == 1) ? kb : vb;

#pragma unroll
  for (int r = 0; r < 8; ++r) {
    int t    = m0 + r + (g << 3);          // global token
    int bimg = t / 196;
    int n    = t - bimg * 196;             // patch index
    int yy   = n / 14, xx = n - (n / 14) * 14;
    float d0 = ac0[r] + bi0;               // dims nn      (h-freqs)
    float d1 = ac1[r] + bi1;               // dims 16+nn   (w-freqs)
    float d2 = ac2[r] + bi2;               // dims 32+nn   (partner of d0)
    float d3 = ac3[r] + bi3;               // dims 48+nn   (partner of d1)
    float o0 = d0, o1 = d1, o2 = d2, o3 = d3;
    if (s < 2) {
      float ah = (float)yy * invf, aw = (float)xx * invf;
      float ch = __cosf(ah), sh = __sinf(ah);
      float cw = __cosf(aw), sw = __sinf(aw);
      o0 = (d0 * ch - d2 * sh) * qscale;
      o2 = (d2 * ch + d0 * sh) * qscale;
      o1 = (d1 * cw - d3 * sw) * qscale;
      o3 = (d3 * cw + d1 * sw) * qscale;
    }
    long base = ((long)(bimg * 12 + h) * 196 + n) * 64 + nn;
    dst[base]      = (__bf16)o0;
    dst[base + 16] = (__bf16)o1;
    dst[base + 32] = (__bf16)o2;
    dst[base + 48] = (__bf16)o3;
  }
}

// ---------------------------------------------------------------------------
// Kernel 2: per (b*h, 32-row q block): Q staged to LDS via the Tensor Data
// Mover, S = QK^T (WMMA), wave softmax with __shfl_xor reductions, P -> bf16
// in LDS, O = P V (WMMA, A frags from LDS). Scores padded to 224 cols so the
// PV K-loop is a multiple of 32.
// ---------------------------------------------------------------------------
#define NPAD 224

__global__ void __launch_bounds__(256) attn_kernel(
    const __bf16* __restrict__ qb, const __bf16* __restrict__ kb,
    const __bf16* __restrict__ vb, __bf16* __restrict__ ob)
{
  __shared__ __bf16 sQ[32 * 64];
  __shared__ float  sS[32 * NPAD];
  __shared__ __bf16 sP[32 * NPAD];

  int lane = threadIdx.x & 31;
  int wv   = threadIdx.x >> 5;
  int bh   = blockIdx.x / 7;
  int row0 = (blockIdx.x % 7) * 32;
  int h    = bh % 12;
  int bimg = bh / 12;
  const __bf16* Q = qb + (long)bh * 196 * 64;
  const __bf16* K = kb + (long)bh * 196 * 64;
  const __bf16* V = vb + (long)bh * 196 * 64;

  // --- TDM: DMA the 32x64 bf16 Q block into LDS. tensor_dim1 = 196-row0 so
  // the hardware zero-fills OOB tail rows. One wave issues; EXEC is ignored
  // by TENSOR ops; completion via TENSORcnt before the workgroup barrier.
  if (wv == 0) {
    unsigned           lds_off = (unsigned)(uintptr_t)(void*)sQ;   // low 32b = LDS byte offset
    unsigned long long ga      = (unsigned long long)(uintptr_t)(Q + (long)row0 * 64);
    unsigned rows_left = (unsigned)(196 - row0);
    v4u g0;
    g0[0] = 1u;                                        // count=1 user descriptor
    g0[1] = lds_off;                                   // lds_addr
    g0[2] = (unsigned)ga;                              // global_addr[31:0]
    g0[3] = ((unsigned)(ga >> 32) & 0x01FFFFFFu) | (2u << 30);  // addr[56:32] | type=2
    v8u g1;
    g1[0] = 0x00010000u;            // data_size = 2 bytes
    g1[1] = (64u << 16);            // tensor_dim0 = 64
    g1[2] = (rows_left << 16);      // tensor_dim1 = remaining rows
    g1[3] = (64u << 16);            // tile_dim0 = 64
    g1[4] = 32u;                    // tile_dim1 = 32
    g1[5] = 64u;                    // tensor_dim0_stride = 64
    g1[6] = 0u;
    g1[7] = 0u;
    asm volatile("tensor_load_to_lds %0, %1" :: "s"(g0), "s"(g1) : "memory");
    __builtin_amdgcn_s_wait_tensorcnt(0);
  }
  __syncthreads();

  // S = Q K^T : 2 row-tiles x 14 col-tiles (padded), round-robin over 8 waves.
  for (int tt = wv; tt < 28; tt += 8) {
    int rt = tt / 14, ct = tt % 14;
    v8f acc = {};
#pragma unroll
    for (int kk = 0; kk < 64; kk += 32) {
      v16bf a = frag_rm_bf16(sQ, 64, rt * 16, kk, lane);               // LDS
      v16bf b = frag_rm_bf16_guard(K, 64, ct * 16, kk, 196, lane);     // K rows = cols of S
      acc = WMMA_BF16(a, b, acc);
    }
    int n = ct * 16 + (lane & 15);
    int g = lane >> 4;
#pragma unroll
    for (int r = 0; r < 8; ++r)
      sS[(rt * 16 + r + (g << 3)) * NPAD + n] = acc[r];
  }
  __syncthreads();

  // Softmax: each wave owns 4 rows; lanes stride the 196 valid columns.
#pragma unroll 1
  for (int rr = 0; rr < 4; ++rr) {
    int r  = wv * 4 + rr;
    int gr = row0 + r;
    float* srow = sS + r * NPAD;
    if (gr < 196) {
      float m = -1e30f;
      for (int c = lane; c < 196; c += 32) m = fmaxf(m, srow[c]);
#pragma unroll
      for (int off = 16; off > 0; off >>= 1) m = fmaxf(m, __shfl_xor(m, off, 32));
      float sum = 0.0f;
      for (int c = lane; c < 196; c += 32) {
        float e = __expf(srow[c] - m);
        srow[c] = e;
        sum += e;
      }
#pragma unroll
      for (int off = 16; off > 0; off >>= 1) sum += __shfl_xor(sum, off, 32);
      float inv = 1.0f / sum;
      for (int c = lane; c < NPAD; c += 32)
        sP[r * NPAD + c] = (c < 196) ? (__bf16)(srow[c] * inv) : (__bf16)0.0f;
    } else {
      for (int c = lane; c < NPAD; c += 32) sP[r * NPAD + c] = (__bf16)0.0f;
    }
  }
  __syncthreads();

  // O = P V : 2 row-tiles x 4 col-tiles = 8 tiles, one per wave.
  {
    int rt = wv >> 2, ct = wv & 3;
    v8f acc = {};
#pragma unroll 1
    for (int kk = 0; kk < NPAD; kk += 32) {
      v16bf a = frag_rm_bf16(sP, NPAD, rt * 16, kk, lane);             // LDS
      v16bf b = frag_kn_bf16_guard(V, 64, kk, ct * 16, 196, lane);     // V is K x N
      acc = WMMA_BF16(a, b, acc);
    }
    int nn = lane & 15;
    int g  = lane >> 4;
#pragma unroll
    for (int r = 0; r < 8; ++r) {
      int gr = row0 + rt * 16 + r + (g << 3);
      if (gr < 196) {
        long t = (long)bimg * 196 + gr;
        ob[t * 768 + h * 64 + ct * 16 + nn] = (__bf16)acc[r];
      }
    }
  }
}

// ---------------------------------------------------------------------------
// Kernel 3: out = attn_out(bf16) @ proj_w^T + proj_b  (fp32 result).
// Each wave computes a 16x64 output slab: 4 accumulators share one A frag.
// ---------------------------------------------------------------------------
__global__ void __launch_bounds__(256) proj_kernel(
    const __bf16* __restrict__ ob, const float* __restrict__ proj_w,
    const float* __restrict__ proj_b, float* __restrict__ out)
{
  int lane = threadIdx.x & 31;
  int wid  = blockIdx.x * (blockDim.x >> 5) + (threadIdx.x >> 5);
  int mt = wid / 12, ng = wid % 12;       // 1568 token tiles x 12 col groups
  int m0 = mt * 16, n0 = ng * 64;

  v8f ac0 = {}, ac1 = {}, ac2 = {}, ac3 = {};
#pragma unroll 1
  for (int kk = 0; kk < 768; kk += 32) {
    if (kk + 32 < 768) {
      __builtin_prefetch(ob     + (long)(m0 + (lane & 15)) * 768 + kk + 32, 0, 3);
      __builtin_prefetch(proj_w + (long)(n0 + (lane & 15)) * 768 + kk + 32, 0, 3);
    }
    v16bf a  = frag_rm_bf16(ob, 768, m0, kk, lane);
    v16bf b0 = frag_rm_f32(proj_w, 768, n0,      kk, lane);
    v16bf b1 = frag_rm_f32(proj_w, 768, n0 + 16, kk, lane);
    v16bf b2 = frag_rm_f32(proj_w, 768, n0 + 32, kk, lane);
    v16bf b3 = frag_rm_f32(proj_w, 768, n0 + 48, kk, lane);
    ac0 = WMMA_BF16(a, b0, ac0);
    ac1 = WMMA_BF16(a, b1, ac1);
    ac2 = WMMA_BF16(a, b2, ac2);
    ac3 = WMMA_BF16(a, b3, ac3);
  }
  int   nn = lane & 15;
  int   g  = lane >> 4;
  float bi0 = proj_b[n0 + nn],      bi1 = proj_b[n0 + 16 + nn];
  float bi2 = proj_b[n0 + 32 + nn], bi3 = proj_b[n0 + 48 + nn];
#pragma unroll
  for (int r = 0; r < 8; ++r) {
    long rowoff = (long)(m0 + r + (g << 3)) * 768 + n0 + nn;
    out[rowoff]      = ac0[r] + bi0;
    out[rowoff + 16] = ac1[r] + bi1;
    out[rowoff + 32] = ac2[r] + bi2;
    out[rowoff + 48] = ac3[r] + bi3;
  }
}

// ---------------------------------------------------------------------------
extern "C" void kernel_launch(void* const* d_in, const int* in_sizes, int n_in,
                              void* d_out, int out_size, void* d_ws, size_t ws_size,
                              hipStream_t stream)
{
  (void)in_sizes; (void)n_in; (void)out_size; (void)ws_size;
  const float* x      = (const float*)d_in[0];
  const float* qkv_w  = (const float*)d_in[1];
  const float* qkv_b  = (const float*)d_in[2];
  const float* proj_w = (const float*)d_in[3];
  const float* proj_b = (const float*)d_in[4];
  float* out = (float*)d_out;

  const long ELEMS = 1536L * 196 * 64;       // per-buffer bf16 element count
  __bf16* qb = (__bf16*)d_ws;                // q (rope'd, pre-scaled)
  __bf16* kb = qb + ELEMS;                   // k (rope'd)
  __bf16* vb = kb + ELEMS;                   // v
  __bf16* ob = vb + ELEMS;                   // attention output [25088 x 768]

  // 1568 token-tiles * 36 (3 sections * 12 heads) waves, 8 waves/block
  qkv_rope_kernel<<<7056, 256, 0, stream>>>(x, qkv_w, qkv_b, qb, kb, vb);
  // 1536 (b*h) * 7 q-blocks of 32 rows
  attn_kernel<<<1536 * 7, 256, 0, stream>>>(qb, kb, vb, ob);
  // 1568 token-tiles * 12 column groups of 64, 8 waves/block
  proj_kernel<<<2352, 256, 0, stream>>>(ob, proj_w, proj_b, out);
}
